// RFCBAMConv_17179869367
// MI455X (gfx1250) — matrix operand order repaired and checked by reading, not compiled, not run on hardware
//
#include <hip/hip_runtime.h>
#include <hip/hip_bf16.h>

// ---------------------------------------------------------------------------
// Types for CDNA5 WMMA
// ---------------------------------------------------------------------------
typedef __attribute__((ext_vector_type(16))) __bf16 v16bf;
typedef __attribute__((ext_vector_type(8)))  float  v8f;

union FragBF16 { unsigned int u[8]; v16bf v; };

#define EPS 1e-5f

static __device__ __forceinline__ unsigned short f2bf(float f) {
    unsigned int u = __float_as_uint(f);
    unsigned int r = u + 0x7FFFu + ((u >> 16) & 1u);   // round-to-nearest-even
    return (unsigned short)(r >> 16);
}
static __device__ __forceinline__ float bf2f(unsigned short h) {
    return __uint_as_float(((unsigned int)h) << 16);
}
static __device__ __forceinline__ float sigmoidf(float x) {
    return 1.0f / (1.0f + __expf(-x));
}

// ---------------------------------------------------------------------------
// Problem constants
// ---------------------------------------------------------------------------
#define B_   8
#define C1_  64
#define C2_  128
#define HIN  256
#define HG   128          // gen-conv output spatial (stride 2)
#define HU   384          // unfolded spatial (HG * 3)
#define HO   192          // final output spatial
#define KTOT 576          // C1 * 9
#define PIX_PER_B (HO*HO) // 36864
#define UPLANE (HU*HU)    // 147456

// ===========================================================================
// Kernel 0: pre-pack cv_w (128,576) f32 -> bf16 (done once per launch)
// ===========================================================================
__global__ void k_pack(const float* __restrict__ cvw,
                       unsigned short* __restrict__ cvwbf) {
    const int i = blockIdx.x * 256 + threadIdx.x;    // 73728 = 288*256
    if (i < C2_ * KTOT) cvwbf[i] = f2bf(cvw[i]);
}

// ===========================================================================
// Kernel 1: global average pool over H,W for each (b,c)
// ===========================================================================
__global__ void k_gap(const float* __restrict__ x, float* __restrict__ gap) {
    __shared__ float red[256];
    const int bc  = blockIdx.x;                 // 0..511
    const int tid = threadIdx.x;
    const float* p = x + (size_t)bc * (HIN * HIN);
    float s = 0.f;
    for (int i = tid; i < HIN * HIN; i += 256) s += p[i];
    red[tid] = s;
    __syncthreads();
    for (int off = 128; off > 0; off >>= 1) {
        if (tid < off) red[tid] += red[tid + off];
        __syncthreads();
    }
    if (tid == 0) gap[bc] = red[0] * (1.0f / (HIN * HIN));
}

// ===========================================================================
// Kernel 2: SE channel attention  ca = sigmoid(relu(gap@W1^T)@W2^T)
// ===========================================================================
__global__ void k_se(const float* __restrict__ gap,
                     const float* __restrict__ w1,   // (16,64)
                     const float* __restrict__ w2,   // (64,16)
                     float* __restrict__ ca) {
    __shared__ float hid[B_ * 16];
    const int tid = threadIdx.x;
    if (tid < B_ * 16) {
        int b = tid >> 4, j = tid & 15;
        float s = 0.f;
        for (int c = 0; c < C1_; ++c) s += gap[b * C1_ + c] * w1[j * C1_ + c];
        hid[tid] = fmaxf(s, 0.f);
    }
    __syncthreads();
    for (int bc = tid; bc < B_ * C1_; bc += blockDim.x) {
        int b = bc >> 6, c = bc & 63;
        float s = 0.f;
        for (int j = 0; j < 16; ++j) s += hid[b * 16 + j] * w2[c * 16 + j];
        ca[bc] = sigmoidf(s);
    }
}

// ===========================================================================
// Kernel 3: depthwise 3x3/s2 gen-conv + BN + ReLU, written directly in the
// unfolded (b,c,384,384) layout as bf16; simultaneously reduces channel
// max and channel sum (-> mean) per unfolded pixel.
// ===========================================================================
__global__ void k_gen(const float* __restrict__ x,
                      const float* __restrict__ gw,     // (576,1,3,3)
                      const float* __restrict__ ggamma,
                      const float* __restrict__ gbeta,
                      const float* __restrict__ gmean,
                      const float* __restrict__ gvar,
                      unsigned short* __restrict__ gf,  // bf16 (b,64,384,384)
                      float* __restrict__ mxb,          // (b,384,384)
                      float* __restrict__ mnb) {
    __shared__ float wlds[KTOT * 9];     // 20.25 KB
    __shared__ float sc[KTOT];
    __shared__ float sh[KTOT];
    const int tid = threadIdx.x;
    for (int i = tid; i < KTOT * 9; i += 256) wlds[i] = gw[i];
    for (int i = tid; i < KTOT; i += 256) {
        float s = ggamma[i] * __frsqrt_rn(gvar[i] + EPS);
        sc[i] = s;
        sh[i] = gbeta[i] - gmean[i] * s;
    }
    __syncthreads();

    const int pos = blockIdx.x * 256 + tid;      // 8*128*128 = 131072
    if (pos >= B_ * HG * HG) return;
    const int b = pos >> 14;
    const int h = (pos >> 7) & 127;
    const int w = pos & 127;

    float xv[3][3];
    float mx9[9], sum9[9];
#pragma unroll
    for (int t = 0; t < 9; ++t) { mx9[t] = -1e30f; sum9[t] = 0.f; }

    const int iy0 = 2 * h - 1, ix0 = 2 * w - 1;

    for (int c = 0; c < C1_; ++c) {
        const float* xp = x + (((size_t)b * C1_ + c) << 16);
#pragma unroll
        for (int i = 0; i < 3; ++i) {
            int iy = iy0 + i;
#pragma unroll
            for (int j = 0; j < 3; ++j) {
                int ix = ix0 + j;
                xv[i][j] = (iy >= 0 && ix >= 0) ? xp[iy * HIN + ix] : 0.f;
            }
        }
        unsigned short* gp = gf + (((size_t)b * C1_ + c) * HU) * HU;
#pragma unroll
        for (int t = 0; t < 9; ++t) {
            const int o = c * 9 + t;
            const float* wt = &wlds[o * 9];
            float a = 0.f;
#pragma unroll
            for (int i = 0; i < 3; ++i)
#pragma unroll
                for (int j = 0; j < 3; ++j)
                    a = fmaf(xv[i][j], wt[i * 3 + j], a);
            float v = fmaxf(a * sc[o] + sh[o], 0.f);
            mx9[t]  = fmaxf(mx9[t], v);
            sum9[t] += v;
            const int y  = 3 * h + t / 3;
            const int x3 = 3 * w + t % 3;
            gp[y * HU + x3] = f2bf(v);
        }
    }
    float* mp = mxb + (size_t)b * UPLANE;
    float* np = mnb + (size_t)b * UPLANE;
#pragma unroll
    for (int t = 0; t < 9; ++t) {
        const int y  = 3 * h + t / 3;
        const int x3 = 3 * w + t % 3;
        mp[y * HU + x3] = mx9[t];
        np[y * HU + x3] = sum9[t] * (1.0f / C1_);
    }
}

// ===========================================================================
// Kernel 4: spatial attention: rfa = sigmoid(conv3x3([mx;mn], sa_w, pad 1))
// ===========================================================================
__global__ void k_rfa(const float* __restrict__ mxb,
                      const float* __restrict__ mnb,
                      const float* __restrict__ saw,   // (1,2,3,3)
                      float* __restrict__ rfa) {
    const int pos = blockIdx.x * 256 + threadIdx.x;   // 8*384*384
    if (pos >= B_ * UPLANE) return;
    const int b = pos / UPLANE;
    const int r = pos - b * UPLANE;
    const int y = r / HU, x = r - y * HU;
    const float* planes[2] = { mxb + (size_t)b * UPLANE, mnb + (size_t)b * UPLANE };
    float a = 0.f;
#pragma unroll
    for (int ch = 0; ch < 2; ++ch) {
        const float* p = planes[ch];
#pragma unroll
        for (int i = 0; i < 3; ++i) {
            int yy = y + i - 1;
            if (yy < 0 || yy >= HU) continue;
#pragma unroll
            for (int j = 0; j < 3; ++j) {
                int xx = x + j - 1;
                if (xx < 0 || xx >= HU) continue;
                a = fmaf(p[yy * HU + xx], saw[ch * 9 + i * 3 + j], a);
            }
        }
    }
    rfa[pos] = sigmoidf(a);
}

// ===========================================================================
// Kernel 5: main conv as implicit GEMM with v_wmma_f32_16x16x32_bf16.
// A-tile (weights, bf16 pre-packed) is brought into LDS with
// global_load_async_to_lds_b64 (ASYNCcnt path); B-tile is the gathered
// gf*ca*rfa product packed to bf16 by the VALU.
// Block: 128 channels x 64 pixels, 8 waves; wave w: M-tile w, N-tiles 0..3.
// ===========================================================================
#define APITCH 36   // ldsA row pitch in halves (72B: 8B-aligned, conflict-free)

__global__ void
__launch_bounds__(256, 2)
k_conv(const unsigned short* __restrict__ gf,
       const float* __restrict__ ca,
       const float* __restrict__ rfa,
       const unsigned short* __restrict__ cvwbf,  // (128,576) bf16
       const float* __restrict__ cgamma,
       const float* __restrict__ cbeta,
       const float* __restrict__ cmean,
       const float* __restrict__ cvar,
       float* __restrict__ out) {
    __shared__ unsigned short ldsA[C2_][APITCH];  // [oc][k]
    __shared__ unsigned short ldsB[64][34];       // [pixel][k]
    __shared__ float cvsc[C2_], cvsh[C2_], cal[C1_];

    const int tid  = threadIdx.x;
    const int b    = blockIdx.x / (PIX_PER_B / 64);
    const int p0   = (blockIdx.x % (PIX_PER_B / 64)) * 64;
    const int wave = tid >> 5;
    const int lane = tid & 31;

    if (tid < C2_) {
        float s = cgamma[tid] * __frsqrt_rn(cvar[tid] + EPS);
        cvsc[tid] = s;
        cvsh[tid] = cbeta[tid] - cmean[tid] * s;
    }
    if (tid < C1_) cal[tid] = ca[b * C1_ + tid];

    v8f acc[4];
#pragma unroll
    for (int t = 0; t < 4; ++t)
#pragma unroll
        for (int r = 0; r < 8; ++r) acc[t][r] = 0.f;

    const unsigned short* gfb = gf + (size_t)b * C1_ * UPLANE;
    const float* rfb = rfa + (size_t)b * UPLANE;

    // LDS byte address base of ldsA for async destinations
    const unsigned ldsAbase = (unsigned)(uintptr_t)&ldsA[0][0];

    // per-thread B-gather geometry (pixel fixed across K chunks)
    int oyb[8], oxb[8], plv[8], klv[8];
#pragma unroll
    for (int i = 0; i < 8; ++i) {
        int e  = tid + 256 * i;          // 2048 elements
        plv[i] = e >> 5;
        klv[i] = e & 31;
        int p  = p0 + plv[i];
        int oy = p / HO;
        oyb[i] = 2 * oy - 1;
        oxb[i] = 2 * (p - oy * HO) - 1;
    }

    for (int kk = 0; kk < KTOT / 32; ++kk) {
        const int k0 = kk * 32;
        __syncthreads();

        // ---- A tile: async copy 128x32 bf16 weights into LDS ------------
        // each lane moves 4 x 8B: row mA = e>>3, 4-half group kg = e&7
#pragma unroll
        for (int i = 0; i < 4; ++i) {
            int e  = tid + 256 * i;                  // 0..1023
            int mA = e >> 3, kg = e & 7;
            unsigned laddr = ldsAbase + (unsigned)(mA * (APITCH * 2) + kg * 8);
            unsigned goff  = (unsigned)((mA * KTOT + k0 + kg * 4) * 2);
            asm volatile("global_load_async_to_lds_b64 %0, %1, %2"
                         :
                         : "v"(laddr), "v"(goff), "s"(cvwbf)
                         : "memory");
        }
        // prefetch next weight chunk while this one is in flight
        if (kk + 1 < KTOT / 32)
            __builtin_prefetch(&cvwbf[(tid >> 1) * KTOT + k0 + 32], 0, 0);

        // ---- B tile: gathered conv_data (64 pix x 32 k) -> bf16 ---------
#pragma unroll
        for (int i = 0; i < 8; ++i) {
            int k  = k0 + klv[i];
            int c  = k / 9;
            int t9 = k - c * 9;
            int ky = t9 / 3, kx = t9 - 3 * (t9 / 3);
            int iy = oyb[i] + ky;
            int ix = oxb[i] + kx;
            float v = 0.f;
            if (iy >= 0 && iy < HU && ix >= 0 && ix < HU) {
                int ui = iy * HU + ix;
                v = bf2f(gfb[(size_t)c * UPLANE + ui]) * cal[c] * rfb[ui];
            }
            ldsB[plv[i]][klv[i]] = f2bf(v);
        }

        // async weight copy must land before the barrier
        asm volatile("s_wait_asynccnt 0x0" ::: "memory");
        __syncthreads();

        // ---- per-wave fragments -----------------------------------------
        // A frag (16-bit 16x32 layout): lanes 0-15 -> K 0..7 & 16..23,
        //                               lanes 16-31 -> K 8..15 & 24..31
        FragBF16 fa, fb[4];
        {
            const unsigned int* arow =
                (const unsigned int*)&ldsA[16 * wave + (lane & 15)][0];
            const int kb2 = (lane >> 4) * 4;      // uint offset (0 or 4)
#pragma unroll
            for (int r = 0; r < 4; ++r) {
                fa.u[r]     = arow[kb2 + r];
                fa.u[4 + r] = arow[kb2 + 8 + r];
            }
        }
#pragma unroll
        for (int t = 0; t < 4; ++t) {
            // B frag (32x16): lanes 0-15 -> K 0..7 & 16..23, lanes 16-31 -> K 8..15 & 24..31
            const unsigned int* brow =
                (const unsigned int*)&ldsB[16 * t + (lane & 15)][0];
            const int kb2 = (lane >> 4) * 4;      // uint offset (0 or 4)
#pragma unroll
            for (int r = 0; r < 4; ++r) {
                fb[t].u[r]     = brow[kb2 + r];
                fb[t].u[4 + r] = brow[kb2 + 8 + r];
            }
        }
        // ---- back-to-back WMMAs (single dscnt wait, XDL pipelining) -----
#pragma unroll
        for (int t = 0; t < 4; ++t)
            acc[t] = __builtin_amdgcn_wmma_f32_16x16x32_bf16(
                false, fa.v, false, fb[t].v, (short)0, acc[t], false, false);
    }
    __syncthreads();

    // ---- epilogue: BN + SiLU, store NCHW --------------------------------
    // D layout: lanes 0-15: VGPR r -> M=r ; lanes 16-31: M=r+8 ; N = lane&15
#pragma unroll
    for (int t = 0; t < 4; ++t) {
        const int p  = p0 + 16 * t + (lane & 15);
        const int mo = (lane >> 4) * 8;
#pragma unroll
        for (int r = 0; r < 8; ++r) {
            const int ch = 16 * wave + mo + r;
            float o = acc[t][r] * cvsc[ch] + cvsh[ch];
            out[((size_t)b * C2_ + ch) * PIX_PER_B + p] = o * sigmoidf(o);
        }
    }
}

// ===========================================================================
// Launch
// ===========================================================================
extern "C" void kernel_launch(void* const* d_in, const int* in_sizes, int n_in,
                              void* d_out, int out_size, void* d_ws, size_t ws_size,
                              hipStream_t stream) {
    const float* x      = (const float*)d_in[0];
    const float* gen_w  = (const float*)d_in[1];
    const float* gen_g  = (const float*)d_in[2];
    const float* gen_b  = (const float*)d_in[3];
    const float* gen_m  = (const float*)d_in[4];
    const float* gen_v  = (const float*)d_in[5];
    const float* se_w1  = (const float*)d_in[6];
    const float* se_w2  = (const float*)d_in[7];
    const float* sa_w   = (const float*)d_in[8];
    const float* cv_w   = (const float*)d_in[9];
    const float* cv_g   = (const float*)d_in[10];
    const float* cv_b   = (const float*)d_in[11];
    const float* cv_m   = (const float*)d_in[12];
    const float* cv_v   = (const float*)d_in[13];
    float* out = (float*)d_out;

    // workspace carve-up
    float* gap = (float*)d_ws;                       // 512
    float* ca  = gap + 512;                          // 512
    float* mxb = ca + 512;                           // 8*384*384
    float* mnb = mxb + (size_t)B_ * UPLANE;
    float* rfa = mnb + (size_t)B_ * UPLANE;
    unsigned short* gf    = (unsigned short*)(rfa + (size_t)B_ * UPLANE);
    unsigned short* cvwbf = gf + (size_t)B_ * C1_ * UPLANE;

    k_pack<<<(C2_ * KTOT) / 256, 256, 0, stream>>>(cv_w, cvwbf);
    k_gap<<<B_ * C1_, 256, 0, stream>>>(x, gap);
    k_se <<<1, 256, 0, stream>>>(gap, se_w1, se_w2, ca);
    k_gen<<<(B_ * HG * HG) / 256, 256, 0, stream>>>(
        x, gen_w, gen_g, gen_b, gen_m, gen_v, gf, mxb, mnb);
    k_rfa<<<(B_ * UPLANE + 255) / 256, 256, 0, stream>>>(mxb, mnb, sa_w, rfa);
    k_conv<<<B_ * (PIX_PER_B / 64), 256, 0, stream>>>(
        gf, ca, rfa, cvwbf, cv_g, cv_b, cv_m, cv_v, out);
}
